// MixingBlock_85152021610852
// MI455X (gfx1250) — matrix operand order repaired and checked by reading, not compiled, not run on hardware
//
#include <hip/hip_runtime.h>

typedef __attribute__((ext_vector_type(16))) _Float16 v16h;
typedef __attribute__((ext_vector_type(8)))  _Float16 v8h;
typedef __attribute__((ext_vector_type(8)))  float    v8f;
typedef __attribute__((ext_vector_type(2)))  int      v2i;

#define DEV __device__ __forceinline__
#define AS1 __attribute__((address_space(1)))
#define AS3 __attribute__((address_space(3)))

// Async global->LDS copy support (gfx1250, ASYNCcnt-tracked). Guarded so the
// file compiles on toolchains that do not declare the builtin.
#if defined(__HIP_DEVICE_COMPILE__) && defined(__gfx1250__) && \
    __has_builtin(__builtin_amdgcn_global_load_async_to_lds_b64)
#define HAS_ASYNC_LDS 1
#else
#define HAS_ASYNC_LDS 0
#endif

DEV void wait_async0() {
#if defined(__HIP_DEVICE_COMPILE__) && __has_builtin(__builtin_amdgcn_s_wait_asynccnt)
    __builtin_amdgcn_s_wait_asynccnt(0);
#elif defined(__HIP_DEVICE_COMPILE__) && defined(__gfx1250__)
    asm volatile("s_wait_asynccnt 0x0" ::: "memory");
#endif
}

// ---------- constants ----------
// DIM=384, c=192, heads=6, hd=32, WS=8, N=64, H=W=128, B=8
// B_ = 2048 windows, M = B_*64 = 131072 token rows = 131072 pixels
static constexpr int MROWS = 131072;

DEV float gelu_f(float x)  { return 0.5f * x * (1.0f + erff(x * 0.70710678118654752f)); }
DEV float sigm_f(float x)  { return 1.0f / (1.0f + __expf(-x)); }
DEV float bn_f(float x, float g, float b, float m, float v) {
    return (x - m) * rsqrtf(v + 1e-5f) * g + b;
}
DEV float wred_sum32(float v) { for (int m = 16; m >= 1; m >>= 1) v += __shfl_xor(v, m); return v; }
DEV v16h joinh(v8h a, v8h b) {
    return __builtin_shufflevector(a, b, 0,1,2,3,4,5,6,7,8,9,10,11,12,13,14,15);
}

enum { EPI_BIAS_F16 = 0, EPI_BIAS_F32 = 1, EPI_LN_F16 = 2, EPI_LN_SCATTER = 3 };

// =====================================================================
// Generic row-strip WMMA GEMM: Out(M x NF) = A(M x K) * W(K x NF) [+ epilogue]
//   - block computes a 32 x NF strip; 256 threads = 8 waves
//   - waves: mt = wave/4 selects 16-row half, wq = wave%4 selects N quarter
//   - A staged 32x32 f16 (async global->LDS when A is f16 row-major),
//     B staged transposed (Bs[n][k]) so the B fragment is a contiguous v16h
//     (lane<16 -> K 0..15, lane>=16 -> K 16..31 same k order as the A fragment)
//   - A_TRANS: A is NCHW f16, A[m][k] = A[(b*K + k)*16384 + hw], m = b*16384+hw
//   - W_TRANS: W is (NF x K) row-major (einsum 'oc' style 1x1 conv weights)
// =====================================================================
template <int NF, int EPI, bool A_F16, bool A_TRANS, bool W_TRANS>
__global__ __launch_bounds__(256) void gemm_rs(
    const void* __restrict__ Ap, const float* __restrict__ Wp,
    const float* __restrict__ bias, const float* __restrict__ gamma,
    const float* __restrict__ beta, void* __restrict__ Outp, int Kdim)
{
    constexpr int NT = NF / 64;                    // wmma N-tiles per wave
    constexpr bool LN = (EPI == EPI_LN_F16 || EPI == EPI_LN_SCATTER);
    constexpr size_t STAGE_B = (size_t)(32 * 32 + NF * 32) * 2;
    constexpr size_t EPIB    = LN ? (size_t)32 * NF * 4 : 0;
    constexpr size_t SB      = (STAGE_B > EPIB) ? STAGE_B : EPIB;
    __shared__ __align__(16) char smem[SB];
    _Float16* As = (_Float16*)smem;                // [32][32]
    _Float16* Bs = (_Float16*)(smem + 32 * 32 * 2);// [NF][32] (n-major)
    float*    Es = (float*)smem;                   // LN scratch [32][NF] (reused)

#if HAS_ASYNC_LDS
    constexpr bool USE_ASYNC = (A_F16 && !A_TRANS);
#else
    constexpr bool USE_ASYNC = false;
#endif

    const int tid  = threadIdx.x;
    const int wave = tid >> 5, lane = tid & 31;
    const int mt = wave >> 2;                      // 0..1
    const int wq = wave & 3;                       // 0..3
    const int nbase = wq * (NF / 4);
    const int m0 = blockIdx.x * 32;
    const int l16 = lane & 15;
    const int khalf = (lane < 16) ? 0 : 16;        // A fragment K base
    const int kb16  = (lane < 16) ? 0 : 16;        // B fragment K base

    v8f acc[NT];
    for (int t = 0; t < NT; ++t)
        for (int i = 0; i < 8; ++i) acc[t][i] = 0.0f;

    for (int k0 = 0; k0 < Kdim; k0 += 32) {
        // ---- stage A 32x32 f16 ----
        if constexpr (USE_ASYNC) {
#if HAS_ASYNC_LDS
            // 256 threads x 8B = exactly the 32x64B tile, ASYNCcnt-tracked DMA
            int r = tid >> 3, chunk = tid & 7;
            _Float16* gp = const_cast<_Float16*>(
                (const _Float16*)Ap + (size_t)(m0 + r) * Kdim + k0 + chunk * 4);
            _Float16* lp = &As[r * 32 + chunk * 4];
            __builtin_amdgcn_global_load_async_to_lds_b64(
                (AS1 v2i*)gp, (AS3 v2i*)lp, 0, 0);
#endif
        } else {
            for (int idx = tid; idx < 32 * 32; idx += 256) {
                int r = idx >> 5, kk = idx & 31;
                int gm = m0 + r, gk = k0 + kk;
                float av;
                if constexpr (A_TRANS) {
                    const _Float16* A16 = (const _Float16*)Ap;
                    int b = gm >> 14, hw = gm & 16383;
                    av = (float)A16[((size_t)b * Kdim + gk) * 16384 + hw];
                } else if constexpr (A_F16) {
                    av = (float)((const _Float16*)Ap)[(size_t)gm * Kdim + gk];
                } else {
                    av = ((const float*)Ap)[(size_t)gm * Kdim + gk];
                }
                As[r * 32 + kk] = (_Float16)av;
            }
        }
        // ---- stage B transposed: Bs[n][kk] (f32 -> f16 convert) ----
        for (int idx = tid; idx < NF * 32; idx += 256) {
            int kk = idx / NF, n = idx % NF;
            float wv;
            if constexpr (W_TRANS) wv = Wp[(size_t)n * Kdim + (k0 + kk)];
            else                   wv = Wp[(size_t)(k0 + kk) * NF + n];
            Bs[n * 32 + kk] = (_Float16)wv;
        }
        if (k0 + 32 < Kdim && !W_TRANS)
            __builtin_prefetch((const char*)&Wp[(size_t)(k0 + 32) * NF], 0, 1);
        if constexpr (USE_ASYNC) wait_async0();
        __syncthreads();

        // ---- fragments + WMMA ----
        const int arow = (mt * 16 + l16) * 32;
        v8h a0 = *(const v8h*)&As[arow + khalf];
        v8h a1 = *(const v8h*)&As[arow + khalf + 8];
        v16h af = joinh(a0, a1);
#pragma unroll
        for (int t = 0; t < NT; ++t) {
            int n = nbase + t * 16 + l16;
            v16h bf = *(const v16h*)&Bs[n * 32 + kb16];
            acc[t] = __builtin_amdgcn_wmma_f32_16x16x32_f16(
                false, af, false, bf, (short)0, acc[t], false, false);
        }
        __syncthreads();
    }

    const int rowoff = (lane >= 16) ? 8 : 0;
    if constexpr (EPI == EPI_BIAS_F16) {
        _Float16* Out = (_Float16*)Outp;
#pragma unroll
        for (int t = 0; t < NT; ++t) {
            int n = nbase + t * 16 + l16; float bv = bias[n];
#pragma unroll
            for (int r = 0; r < 8; ++r) {
                int gm = m0 + mt * 16 + r + rowoff;
                Out[(size_t)gm * NF + n] = (_Float16)(acc[t][r] + bv);
            }
        }
    } else if constexpr (EPI == EPI_BIAS_F32) {
        float* Out = (float*)Outp;
#pragma unroll
        for (int t = 0; t < NT; ++t) {
            int n = nbase + t * 16 + l16; float bv = bias[n];
#pragma unroll
            for (int r = 0; r < 8; ++r) {
                int gm = m0 + mt * 16 + r + rowoff;
                Out[(size_t)gm * NF + n] = acc[t][r] + bv;
            }
        }
    } else {
        // LayerNorm epilogue: z = xW + b dumped to LDS, then rowwise LN
#pragma unroll
        for (int t = 0; t < NT; ++t) {
            int n = nbase + t * 16 + l16; float bv = bias[n];
#pragma unroll
            for (int r = 0; r < 8; ++r) {
                int lm = mt * 16 + r + rowoff;
                Es[lm * NF + n] = acc[t][r] + bv;
            }
        }
        __syncthreads();
        constexpr int NL = NF / 32;
        for (int j = 0; j < 4; ++j) {                 // 4 rows per wave
            int lr = wave * 4 + j;
            int gm = m0 + lr;
            float v[NL]; float s = 0.0f;
#pragma unroll
            for (int q = 0; q < NL; ++q) { v[q] = Es[lr * NF + lane + 32 * q]; s += v[q]; }
            s = wred_sum32(s);
            float mean = s * (1.0f / NF);
            float qs = 0.0f;
#pragma unroll
            for (int q = 0; q < NL; ++q) { float d = v[q] - mean; qs += d * d; }
            qs = wred_sum32(qs);
            float rs = rsqrtf(qs * (1.0f / NF) + 1e-5f);
#pragma unroll
            for (int q = 0; q < NL; ++q) {
                int n = lane + 32 * q;
                float y = (v[q] - mean) * rs * gamma[n] + beta[n];
                if constexpr (EPI == EPI_LN_F16) {
                    ((_Float16*)Outp)[(size_t)gm * NF + n] = (_Float16)y;
                } else {
                    // window_reverse2 scatter -> NCHW (Bi, NF, 128, 128)
                    int w = gm >> 6, nn = gm & 63;
                    int bi = w >> 8, hb = (w >> 4) & 15, wb = w & 15;
                    int h = hb * 8 + (nn >> 3), xx = wb * 8 + (nn & 7);
                    ((_Float16*)Outp)[(((size_t)bi * NF + n) << 14) + (h << 7) + xx] = (_Float16)y;
                }
            }
        }
    }
}

// =====================================================================
// Precompute relative-position bias table: biasT[head][query m][key n]
// (6 x 64 x 64 f32; removes 12288 x 4096 redundant gathers from rpb)
// =====================================================================
__global__ __launch_bounds__(256) void bias_kernel(const float* __restrict__ rpb,
                                                   float* __restrict__ biasT)
{
    int idx = blockIdx.x * 256 + threadIdx.x;      // 6*64*64 = 24576
    int head = idx >> 12, mm = (idx >> 6) & 63, nn = idx & 63;
    int ridx = ((mm >> 3) - (nn >> 3) + 7) * 15 + ((mm & 7) - (nn & 7) + 7);
    biasT[idx] = rpb[ridx * 6 + head];
}

// =====================================================================
// Windowed attention: one block (128 thr, 4 waves) per (head, window).
// S = (q*scale) k^T + bias, row softmax, O = P (v*gate). All WMMA.
// =====================================================================
__global__ __launch_bounds__(128) void attn_kernel(
    const _Float16* __restrict__ qkv, const float* __restrict__ biasT,
    const float* __restrict__ gate, _Float16* __restrict__ xaout)
{
    const int head = blockIdx.x;
    const int w    = blockIdx.y;
    const int tid = threadIdx.x, wv = tid >> 5, lane = tid & 31;
    const int l16 = lane & 15;
    const int khalf = (lane < 16) ? 0 : 16;
    const int kb16  = (lane < 16) ? 0 : 16;
    const int rowoff = (lane >= 16) ? 8 : 0;
    __shared__ __align__(16) _Float16 Pl[64 * 64];   // probs, row-major
    __shared__ __align__(16) _Float16 VgT[32 * 64];  // gated V, transposed [t][n]
    const int bi = w >> 8;
    const size_t base = (size_t)w * 64 * 576;
    const float* bh = biasT + head * 4096;

    // stage gated V transposed
    for (int idx = tid; idx < 64 * 32; idx += 128) {
        int n = idx >> 5, t = idx & 31;
        float vv = (float)qkv[base + (size_t)n * 576 + 384 + head * 32 + t];
        VgT[t * 64 + n] = (_Float16)(vv * gate[bi * 192 + head * 32 + t]);
    }

    // ---- S = q k^T (M=64 split over waves, N=64, K=32) ----
    const _Float16* qrow = qkv + base + (size_t)(wv * 16 + l16) * 576 + head * 32;
    v8h a0 = *(const v8h*)(qrow + khalf);
    v8h a1 = *(const v8h*)(qrow + khalf + 8);
    v16h af = joinh(a0, a1);
    v8f sacc[4];
    for (int t = 0; t < 4; ++t) for (int i = 0; i < 8; ++i) sacc[t][i] = 0.0f;
#pragma unroll
    for (int nt = 0; nt < 4; ++nt) {
        const _Float16* krow = qkv + base + (size_t)(nt * 16 + l16) * 576 + 192 + head * 32;
        v16h bf = *(const v16h*)(krow + kb16);
        sacc[nt] = __builtin_amdgcn_wmma_f32_16x16x32_f16(
            false, af, false, bf, (short)0, sacc[nt], false, false);
    }
    // scale + relative position bias (coalesced from precomputed table)
    const float scale = 0.17677669529663689f;  // 32^-0.5
#pragma unroll
    for (int nt = 0; nt < 4; ++nt)
#pragma unroll
        for (int r = 0; r < 8; ++r) {
            int mm = wv * 16 + r + rowoff, nn = nt * 16 + l16;
            sacc[nt][r] = sacc[nt][r] * scale + bh[mm * 64 + nn];
        }
    // row softmax (row = fixed r, spread over 16 lanes x 4 ntiles)
#pragma unroll
    for (int r = 0; r < 8; ++r) {
        float mx = sacc[0][r];
#pragma unroll
        for (int nt = 1; nt < 4; ++nt) mx = fmaxf(mx, sacc[nt][r]);
        for (int m = 8; m >= 1; m >>= 1) mx = fmaxf(mx, __shfl_xor(mx, m));
        float s = 0.0f;
#pragma unroll
        for (int nt = 0; nt < 4; ++nt) { float e = __expf(sacc[nt][r] - mx); sacc[nt][r] = e; s += e; }
        for (int m = 8; m >= 1; m >>= 1) s += __shfl_xor(s, m);
        float inv = 1.0f / s;
        int mm = wv * 16 + r + rowoff;
#pragma unroll
        for (int nt = 0; nt < 4; ++nt)
            Pl[mm * 64 + nt * 16 + l16] = (_Float16)(sacc[nt][r] * inv);
    }
    __syncthreads();

    // ---- O = P @ Vg (M=16/wave, N=32, K=64) ----
    v8f oacc[2];
    for (int t = 0; t < 2; ++t) for (int i = 0; i < 8; ++i) oacc[t][i] = 0.0f;
#pragma unroll
    for (int ks = 0; ks < 64; ks += 32) {
        const _Float16* prow = &Pl[(wv * 16 + l16) * 64 + ks + khalf];
        v8h p0 = *(const v8h*)prow;
        v8h p1 = *(const v8h*)(prow + 8);
        v16h pf = joinh(p0, p1);
#pragma unroll
        for (int nt = 0; nt < 2; ++nt) {
            int t = nt * 16 + l16;
            v16h vf = *(const v16h*)&VgT[t * 64 + ks + kb16];
            oacc[nt] = __builtin_amdgcn_wmma_f32_16x16x32_f16(
                false, pf, false, vf, (short)0, oacc[nt], false, false);
        }
    }
#pragma unroll
    for (int nt = 0; nt < 2; ++nt)
#pragma unroll
        for (int r = 0; r < 8; ++r) {
            int mm = wv * 16 + r + rowoff, tt = nt * 16 + l16;
            xaout[(size_t)(w * 64 + mm) * 192 + head * 32 + tt] = (_Float16)oacc[nt][r];
        }
}

// =====================================================================
// depthwise 3x3 conv + bias + BN + exact GELU, NCHW f16 in/out
// =====================================================================
__global__ __launch_bounds__(256) void dwconv_kernel(
    const _Float16* __restrict__ xin, const float* __restrict__ wgt,
    const float* __restrict__ cb, const float* __restrict__ g,
    const float* __restrict__ be, const float* __restrict__ mn,
    const float* __restrict__ vr, _Float16* __restrict__ xout)
{
    size_t idx = (size_t)blockIdx.x * 256 + threadIdx.x;   // 8*384*16384 total
    int xx = (int)(idx & 127), h = (int)((idx >> 7) & 127);
    size_t bc = idx >> 14;
    int ch = (int)(bc % 384);
    const _Float16* plane = xin + (bc << 14);
    float a = 0.0f;
#pragma unroll
    for (int ky = 0; ky < 3; ++ky) {
        int hy = h + ky - 1; if ((unsigned)hy >= 128u) continue;
#pragma unroll
        for (int kx = 0; kx < 3; ++kx) {
            int hx = xx + kx - 1; if ((unsigned)hx >= 128u) continue;
            a += (float)plane[hy * 128 + hx] * wgt[ch * 9 + ky * 3 + kx];
        }
    }
    a += cb[ch];
    a = bn_f(a, g[ch], be[ch], mn[ch], vr[ch]);
    xout[idx] = (_Float16)gelu_f(a);
}

// per-(b,channel) spatial mean of xc1 -> (8,384)
__global__ __launch_bounds__(256) void mean_kernel(const _Float16* __restrict__ xc1,
                                                   float* __restrict__ meanb)
{
    int bc = blockIdx.x;                     // 0..3071
    const _Float16* p = xc1 + ((size_t)bc << 14);
    float s = 0.0f;
    for (int i = threadIdx.x; i < 16384; i += 256) s += (float)p[i];
    s = wred_sum32(s);
    __shared__ float wsum[8];
    if ((threadIdx.x & 31) == 0) wsum[threadIdx.x >> 5] = s;
    __syncthreads();
    if (threadIdx.x == 0) {
        float t = 0.0f;
        for (int i = 0; i < 8; ++i) t += wsum[i];
        meanb[bc] = t * (1.0f / 16384.0f);
    }
}

// SE channel interaction: gate = sigmoid(w2 @ gelu(bn(w1 @ mean + b1)) + b2)
__global__ __launch_bounds__(256) void se_kernel(
    const float* __restrict__ meanb, const float* __restrict__ w1,
    const float* __restrict__ b1, const float* __restrict__ g,
    const float* __restrict__ be, const float* __restrict__ mn,
    const float* __restrict__ vr, const float* __restrict__ w2,
    const float* __restrict__ b2, float* __restrict__ gateb)
{
    int b = blockIdx.x;
    __shared__ float t1[48];
    const float* m_ = meanb + b * 384;
    int tid = threadIdx.x;
    if (tid < 48) {
        float s = b1[tid];
        for (int k = 0; k < 384; ++k) s += w1[tid * 384 + k] * m_[k];
        t1[tid] = gelu_f(bn_f(s, g[tid], be[tid], mn[tid], vr[tid]));
    }
    __syncthreads();
    if (tid < 192) {
        float s = b2[tid];
        for (int k = 0; k < 48; ++k) s += w2[tid * 48 + k] * t1[k];
        gateb[b * 192 + tid] = sigm_f(s);
    }
}

// spatial interaction: si = sigmoid(w2 @ gelu(bn(w1 @ xs + b1)) + b2), per pixel
__global__ __launch_bounds__(256) void si_kernel(
    const _Float16* __restrict__ xa, const float* __restrict__ w1,
    const float* __restrict__ b1, const float* __restrict__ g,
    const float* __restrict__ be, const float* __restrict__ mn,
    const float* __restrict__ vr, const float* __restrict__ w2,
    const float* __restrict__ b2, float* __restrict__ sib)
{
    size_t idx = (size_t)blockIdx.x * 256 + threadIdx.x;   // 131072 pixels
    int b = (int)(idx >> 14), hw = (int)(idx & 16383);
    int h = hw >> 7, xx = hw & 127;
    int w = b * 256 + (h >> 3) * 16 + (xx >> 3);
    int nn = (h & 7) * 8 + (xx & 7);
    const _Float16* row = xa + (size_t)(w * 64 + nn) * 192;
    float s[48];
#pragma unroll
    for (int o = 0; o < 48; ++o) s[o] = b1[o];
    for (int k = 0; k < 192; ++k) {
        float xv = (float)row[k];
#pragma unroll
        for (int o = 0; o < 48; ++o) s[o] += w1[o * 192 + k] * xv;
    }
    float acc = b2[0];
#pragma unroll
    for (int o = 0; o < 48; ++o)
        acc += w2[o] * gelu_f(bn_f(s[o], g[o], be[o], mn[o], vr[o]));
    sib[idx] = sigm_f(acc);
}

// LN(xa) -> concat columns [0,192)
__global__ __launch_bounds__(256) void lnxa_kernel(
    const _Float16* __restrict__ xa, const float* __restrict__ g,
    const float* __restrict__ be, _Float16* __restrict__ catb)
{
    int row = blockIdx.x * 8 + (threadIdx.x >> 5);
    int lane = threadIdx.x & 31;
    const _Float16* p = xa + (size_t)row * 192;
    float v[6]; float s = 0.0f;
#pragma unroll
    for (int j = 0; j < 6; ++j) { v[j] = (float)p[lane + 32 * j]; s += v[j]; }
    s = wred_sum32(s);
    float mean = s * (1.0f / 192.0f);
    float qs = 0.0f;
#pragma unroll
    for (int j = 0; j < 6; ++j) { float d = v[j] - mean; qs += d * d; }
    qs = wred_sum32(qs);
    float rs = rsqrtf(qs * (1.0f / 192.0f) + 1e-5f);
#pragma unroll
    for (int j = 0; j < 6; ++j) {
        int n = lane + 32 * j;
        catb[(size_t)row * 384 + n] = (_Float16)((v[j] - mean) * rs * g[n] + be[n]);
    }
}

// sigmoid(si) * xc2, BN(cn), window_partition2 -> concat columns [192,384)
__global__ __launch_bounds__(256) void pack_kernel(
    const _Float16* __restrict__ xc2, const float* __restrict__ sib,
    const float* __restrict__ g, const float* __restrict__ be,
    const float* __restrict__ mn, const float* __restrict__ vr,
    _Float16* __restrict__ catb)
{
    size_t idx = (size_t)blockIdx.x * 256 + threadIdx.x;   // 131072*192
    int ch = (int)(idx % 192);
    size_t pix = idx / 192;
    float v = (float)xc2[idx] * sib[pix];
    v = bn_f(v, g[ch], be[ch], mn[ch], vr[ch]);
    int b = (int)(pix >> 14), hw = (int)(pix & 16383);
    int h = hw >> 7, xx = hw & 127;
    int w = b * 256 + (h >> 3) * 16 + (xx >> 3);
    int nn = (h & 7) * 8 + (xx & 7);
    catb[(size_t)(w * 64 + nn) * 384 + 192 + ch] = (_Float16)v;
}

// =====================================================================
extern "C" void kernel_launch(void* const* d_in, const int* in_sizes, int n_in,
                              void* d_out, int out_size, void* d_ws, size_t ws_size,
                              hipStream_t stream)
{
    (void)in_sizes; (void)n_in; (void)out_size; (void)ws_size;
    const float* x     = (const float*)d_in[0];
    const float* w_pa  = (const float*)d_in[1];
    const float* b_pa  = (const float*)d_in[2];
    const float* g_pan = (const float*)d_in[3];
    const float* be_pan= (const float*)d_in[4];
    const float* w_pc  = (const float*)d_in[5];
    const float* b_pc  = (const float*)d_in[6];
    const float* g_pcn = (const float*)d_in[7];
    const float* be_pcn= (const float*)d_in[8];
    const float* dw_w  = (const float*)d_in[9];
    const float* dw_b  = (const float*)d_in[10];
    const float* dw_g  = (const float*)d_in[11];
    const float* dw_be = (const float*)d_in[12];
    const float* dw_m  = (const float*)d_in[13];
    const float* dw_v  = (const float*)d_in[14];
    const float* ci_w1 = (const float*)d_in[15];
    const float* ci_b1 = (const float*)d_in[16];
    const float* ci_g  = (const float*)d_in[17];
    const float* ci_be = (const float*)d_in[18];
    const float* ci_m  = (const float*)d_in[19];
    const float* ci_v  = (const float*)d_in[20];
    const float* ci_w2 = (const float*)d_in[21];
    const float* ci_b2 = (const float*)d_in[22];
    const float* pj_w  = (const float*)d_in[23];
    const float* pj_b  = (const float*)d_in[24];
    const float* cn_g  = (const float*)d_in[25];
    const float* cn_be = (const float*)d_in[26];
    const float* cn_m  = (const float*)d_in[27];
    const float* cn_v  = (const float*)d_in[28];
    const float* qkv_w = (const float*)d_in[29];
    const float* qkv_b = (const float*)d_in[30];
    const float* rpb   = (const float*)d_in[31];
    const float* si_w1 = (const float*)d_in[32];
    const float* si_b1 = (const float*)d_in[33];
    const float* si_g  = (const float*)d_in[34];
    const float* si_be = (const float*)d_in[35];
    const float* si_m  = (const float*)d_in[36];
    const float* si_v  = (const float*)d_in[37];
    const float* si_w2 = (const float*)d_in[38];
    const float* si_b2 = (const float*)d_in[39];
    const float* an_g  = (const float*)d_in[40];
    const float* an_b  = (const float*)d_in[41];
    const float* pr_w  = (const float*)d_in[42];
    const float* pr_b  = (const float*)d_in[43];

    char* ws = (char*)d_ws;
    size_t off = 0;
    auto take = [&](size_t bytes) -> char* {
        char* p = ws + off;
        off += (bytes + 255) & ~(size_t)255;
        return p;
    };
    _Float16* xa16  = (_Float16*)take((size_t)MROWS * 192 * 2);     // LN proj (atten)
    _Float16* xcnn  = (_Float16*)take((size_t)8 * 384 * 16384 * 2); // conv input NCHW
    _Float16* xc1   = (_Float16*)take((size_t)8 * 384 * 16384 * 2); // conv+GELU NCHW
    _Float16* qkvb  = (_Float16*)take((size_t)MROWS * 576 * 2);
    _Float16* xaout = (_Float16*)take((size_t)MROWS * 192 * 2);     // attention out
    _Float16* xc2   = (_Float16*)take((size_t)MROWS * 192 * 2);     // pj out, pixel-major
    _Float16* catb  = (_Float16*)take((size_t)MROWS * 384 * 2);     // concat input
    float* sib   = (float*)take((size_t)MROWS * 4);
    float* meanb = (float*)take((size_t)3072 * 4);
    float* gateb = (float*)take((size_t)1536 * 4);
    float* biasT = (float*)take((size_t)6 * 64 * 64 * 4);
    float* outf  = (float*)d_out;

    const int GEMM_GRID = MROWS / 32;   // 4096

    // 0) bias table (tiny, stays in L2)
    bias_kernel<<<96, 256, 0, stream>>>(rpb, biasT);

    // 1) projections (fp32 A, K=384)
    gemm_rs<192, EPI_LN_F16,    false, false, false><<<GEMM_GRID, 256, 0, stream>>>(
        (const void*)x, w_pa, b_pa, g_pan, be_pan, (void*)xa16, 384);
    gemm_rs<384, EPI_LN_SCATTER,false, false, false><<<GEMM_GRID, 256, 0, stream>>>(
        (const void*)x, w_pc, b_pc, g_pcn, be_pcn, (void*)xcnn, 384);

    // 2) conv branch
    dwconv_kernel<<<196608, 256, 0, stream>>>(xcnn, dw_w, dw_b, dw_g, dw_be, dw_m, dw_v, xc1);
    mean_kernel<<<3072, 256, 0, stream>>>(xc1, meanb);
    se_kernel<<<8, 256, 0, stream>>>(meanb, ci_w1, ci_b1, ci_g, ci_be, ci_m, ci_v,
                                     ci_w2, ci_b2, gateb);

    // 3) attention branch (f16 A -> async global->LDS staging)
    gemm_rs<576, EPI_BIAS_F16, true, false, false><<<GEMM_GRID, 256, 0, stream>>>(
        (const void*)xa16, qkv_w, qkv_b, nullptr, nullptr, (void*)qkvb, 192);
    attn_kernel<<<dim3(6, 2048), 128, 0, stream>>>(qkvb, biasT, gateb, xaout);

    // 4) pj 1x1 (A = xc1 NCHW gather, W transposed (192x384))
    gemm_rs<192, EPI_BIAS_F16, true, true, true><<<GEMM_GRID, 256, 0, stream>>>(
        (const void*)xc1, pj_w, pj_b, nullptr, nullptr, (void*)xc2, 384);

    // 5) spatial gate + pack both halves of concat
    si_kernel<<<512, 256, 0, stream>>>(xaout, si_w1, si_b1, si_g, si_be, si_m, si_v,
                                       si_w2, si_b2, sib);
    lnxa_kernel<<<16384, 256, 0, stream>>>(xaout, an_g, an_b, catb);
    pack_kernel<<<98304, 256, 0, stream>>>(xc2, sib, cn_g, cn_be, cn_m, cn_v, catb);

    // 6) final projection -> fp32 output (f16 A -> async staging)
    gemm_rs<384, EPI_BIAS_F32, true, false, false><<<GEMM_GRID, 256, 0, stream>>>(
        (const void*)catb, pr_w, pr_b, nullptr, nullptr, (void*)outf, 384);
}
